// GNN_23665269801390
// MI455X (gfx1250) — matrix operand order repaired and checked by reading, not compiled, not run on hardware
//
#include <hip/hip_runtime.h>
#include <hip/hip_bf16.h>

#define N_NODES 100000
#define N_EDGES 1600000
#define DFEAT   128
#define NLAYERS 5
#define BN_EPS  1e-5f

typedef _Float16 half16 __attribute__((ext_vector_type(16)));
typedef _Float16 half8  __attribute__((ext_vector_type(8)));
typedef float    float8 __attribute__((ext_vector_type(8)));
typedef float    floatx4 __attribute__((ext_vector_type(4)));

union AOp { half16 v; half8 h[2]; };

// ---------------------------------------------------------------------------
// Weight pre-pack: f32 row-major W -> f16 transposed (col-major) for B-lane
// contiguous loads.  W1:[L][128][256] -> W1t:[L][256][128]
//                    W2:[L][256][128] -> W2t:[L][128][256]
// ---------------------------------------------------------------------------
__global__ __launch_bounds__(256) void pack_weights(
    const float* __restrict__ W1, const float* __restrict__ W2,
    _Float16* __restrict__ W1t, _Float16* __restrict__ W2t)
{
    int idx = blockIdx.x * 256 + threadIdx.x;
    const int T = NLAYERS * 128 * 256;
    if (idx < T) {
        int l = idx / (128 * 256);
        int rem = idx % (128 * 256);
        int k = rem / 256;          // input dim 0..127
        int n = rem % 256;          // output col 0..255
        W1t[(size_t)l * 256 * 128 + (size_t)n * 128 + k] = (_Float16)W1[idx];
    } else {
        int j = idx - T;
        if (j < T) {
            int l = j / (256 * 128);
            int rem = j % (256 * 128);
            int k = rem / 128;      // input dim 0..255
            int n = rem % 128;      // output col 0..127
            W2t[(size_t)l * 128 * 256 + (size_t)n * 256 + k] = (_Float16)W2[j];
        }
    }
}

// ---------------------------------------------------------------------------
// Node embedding: sum of 7 table lookups -> h [N,128] f32
// ---------------------------------------------------------------------------
__global__ __launch_bounds__(256) void node_embed(
    const int* __restrict__ x,
    const float* __restrict__ ta, const float* __restrict__ td,
    const float* __restrict__ tf, const float* __restrict__ tc,
    const float* __restrict__ th_, const float* __restrict__ tn,
    const float* __restrict__ ti, float* __restrict__ h)
{
    int idx = blockIdx.x * 256 + threadIdx.x;     // N*128 threads exactly
    int node = idx >> 7;
    int d = idx & 127;
    const int* xr = x + node * 7;
    h[idx] = ta[xr[0] * DFEAT + d] + td[xr[1] * DFEAT + d]
           + tf[xr[2] * DFEAT + d] + tc[xr[3] * DFEAT + d]
           + th_[xr[4] * DFEAT + d] + tn[xr[5] * DFEAT + d]
           + ti[xr[6] * DFEAT + d];
}

// ---------------------------------------------------------------------------
// Message scatter: one wave per edge, float4 per lane.
// agg[dst] += h[src] + e1[a0] + e2[a1] + e3[a2] + e4[a3]
// h (51 MB) is L2-resident on MI455X (192 MB L2) -> random gather stays on-chip.
// ---------------------------------------------------------------------------
__global__ __launch_bounds__(256) void scatter_msgs(
    const float* __restrict__ h, const int* __restrict__ ei,
    const int* __restrict__ ea,
    const float* __restrict__ t1, const float* __restrict__ t2,
    const float* __restrict__ t3, const float* __restrict__ t4,
    float* __restrict__ agg)
{
    int edge = blockIdx.x * 8 + (threadIdx.x >> 5);   // E divisible by 8
    int lane = threadIdx.x & 31;
    int src = ei[edge];
    int dst = ei[N_EDGES + edge];
    int a0 = ea[edge * 4 + 0], a1 = ea[edge * 4 + 1];
    int a2 = ea[edge * 4 + 2], a3 = ea[edge * 4 + 3];
    int d = lane * 4;
    floatx4 m = *(const floatx4*)(h + (size_t)src * DFEAT + d);
    m += *(const floatx4*)(t1 + (size_t)a0 * DFEAT + d);
    m += *(const floatx4*)(t2 + (size_t)a1 * DFEAT + d);
    m += *(const floatx4*)(t3 + (size_t)a2 * DFEAT + d);
    m += *(const floatx4*)(t4 + (size_t)a3 * DFEAT + d);
    float* ap = agg + (size_t)dst * DFEAT + d;
    atomicAdd(ap + 0, m.x);
    atomicAdd(ap + 1, m.y);
    atomicAdd(ap + 2, m.z);
    atomicAdd(ap + 3, m.w);
}

// ---------------------------------------------------------------------------
// Fold self-loop + convert to f16: X16 = f16(agg + h + ee_self)
// ee_self = e1[4] + e2[0] + e3[0] + e4[0]  (same vector for every node)
// ---------------------------------------------------------------------------
__global__ __launch_bounds__(256) void combine_x16(
    const float* __restrict__ agg, const float* __restrict__ h,
    const float* __restrict__ t1, const float* __restrict__ t2,
    const float* __restrict__ t3, const float* __restrict__ t4,
    _Float16* __restrict__ X16)
{
    int idx = blockIdx.x * 256 + threadIdx.x;
    int d = idx & 127;
    float ee = t1[4 * DFEAT + d] + t2[d] + t3[d] + t4[d];
    X16[idx] = (_Float16)(agg[idx] + h[idx] + ee);
}

// ---------------------------------------------------------------------------
// WMMA GEMM:  out[nrows,NOUT] = X[nrows,K] @ Wt^T + bias
//   Wt is pre-transposed f16 [NOUT][K].
//   Block = 256 threads = 8 waves; block covers (8*RT*16) rows x (CT*16) cols.
//   Wave w owns RT 16-row tiles x CT 16-col tiles -> RT*CT WMMAs / k-step.
//   All RT A-tiles and CT B-tiles are fetched into distinct registers before
//   the WMMA burst so ds/global loads overlap and WMMAs issue back-to-back.
//   B panel staged through LDS in the documented wave32 per-lane layout.
//   RELU_STORE_F16: fuse ReLU and emit f16 (feeds next GEMM); else f32.
// ---------------------------------------------------------------------------
template<int K, int NOUT, int RT, int CT, bool RELU_STORE_F16>
__global__ __launch_bounds__(256) void gemm_wmma(
    const _Float16* __restrict__ X, const _Float16* __restrict__ Wt,
    const float* __restrict__ bias,
    _Float16* __restrict__ out16, float* __restrict__ out32, int nrows)
{
    constexpr int KT = K / 32;                       // k-steps of 32
    __shared__ __align__(16) _Float16 ldsB[CT * KT * 32 * 16];

    const int tid = threadIdx.x;
    const int colbase = blockIdx.y * (CT * 16);

    // Cooperative B-panel load into LDS, packed so each lane's 16 halves are
    // contiguous:  entry(c,t,lane) holds col = colbase+16c+(lane&15),
    // K = 32t + 8*(lane>>4) + 0..7, then 32t + 16 + 8*(lane>>4) + 0..7   (ISA 05_wmma B layout, wave32)
    for (int e = tid; e < CT * KT * 32; e += 256) {
        int lane = e & 31;
        int t    = (e >> 5) % KT;
        int c    = e / (KT * 32);
        int col  = colbase + c * 16 + (lane & 15);
        int k0   = t * 32 + ((lane >> 4) << 3);
        const _Float16* src = Wt + (size_t)col * K + k0;
        _Float16* dst = &ldsB[(size_t)((c * KT + t) * 32 + lane) * 16];
        *(half8*)(dst)     = *(const half8*)(src);
        *(half8*)(dst + 8) = *(const half8*)(src + 16);
    }
    __syncthreads();

    const int wave = tid >> 5;
    const int lane = tid & 31;
    const int row0 = blockIdx.x * (8 * RT * 16) + wave * (RT * 16);
    if (row0 >= nrows) return;    // uniform per wave; EXEC stays all-1 for WMMA

    float8 acc[RT][CT];
    #pragma unroll
    for (int rt = 0; rt < RT; ++rt)
        #pragma unroll
        for (int c = 0; c < CT; ++c) acc[rt][c] = (float8){};

    // A layout (16x32 f16): lane holds row M = lane&15; halves 0..7 are
    // K = 8*(lane>>4)+0..7, halves 8..15 are K = 8*(lane>>4)+16..23.
    const _Float16* aptr[RT];
    #pragma unroll
    for (int rt = 0; rt < RT; ++rt)
        aptr[rt] = X + (size_t)(row0 + rt * 16 + (lane & 15)) * K
                     + ((lane >> 4) << 3);

    #pragma unroll
    for (int t = 0; t < KT; ++t) {
        AOp a[RT];
        #pragma unroll
        for (int rt = 0; rt < RT; ++rt) {
            a[rt].h[0] = *(const half8*)(aptr[rt]);
            a[rt].h[1] = *(const half8*)(aptr[rt] + 16);
            __builtin_prefetch(aptr[rt] + 32, 0, 0);  // global_prefetch_b8
            aptr[rt] += 32;
        }
        AOp b[CT];
        #pragma unroll
        for (int c = 0; c < CT; ++c) {
            const _Float16* bp = &ldsB[(size_t)((c * KT + t) * 32 + lane) * 16];
            b[c].h[0] = *(const half8*)(bp);
            b[c].h[1] = *(const half8*)(bp + 8);
        }
        #pragma unroll
        for (int c = 0; c < CT; ++c)
            #pragma unroll
            for (int rt = 0; rt < RT; ++rt)
                acc[rt][c] = __builtin_amdgcn_wmma_f32_16x16x32_f16(
                    false, a[rt].v, false, b[c].v, (short)0,
                    acc[rt][c], false, false);
    }

    // C/D layout: lane -> col = (lane&15); VGPR r -> row = rtile0 + 8*(lane>>4) + r
    #pragma unroll
    for (int rt = 0; rt < RT; ++rt) {
        const int rbase = row0 + rt * 16 + ((lane >> 4) << 3);
        #pragma unroll
        for (int c = 0; c < CT; ++c) {
            const int col = colbase + c * 16 + (lane & 15);
            const float bv = bias[col];
            #pragma unroll
            for (int r = 0; r < 8; ++r) {
                float v = acc[rt][c][r] + bv;
                size_t ro = (size_t)(rbase + r) * NOUT + col;
                if constexpr (RELU_STORE_F16) {
                    out16[ro] = (_Float16)fmaxf(v, 0.f);
                } else {
                    out32[ro] = v;
                }
            }
        }
    }
}

// ---------------------------------------------------------------------------
// BatchNorm stats: per-column sum & sum-of-squares over N rows.
// Two-level: LDS partials -> 256 global atomics per block.
// ---------------------------------------------------------------------------
__global__ __launch_bounds__(256) void bn_stats(
    const float* __restrict__ h2, float* __restrict__ stats, int nrows)
{
    __shared__ float shs[256], shq[256];
    int tid = threadIdx.x;
    int col = tid & 127;
    int rsub = tid >> 7;
    float s = 0.f, q = 0.f;
    for (int r = blockIdx.x * 2 + rsub; r < nrows; r += gridDim.x * 2) {
        float v = h2[(size_t)r * DFEAT + col];
        s += v; q += v * v;
    }
    shs[tid] = s; shq[tid] = q;
    __syncthreads();
    if (tid < 128) {
        atomicAdd(&stats[col],       shs[tid] + shs[tid + 128]);
        atomicAdd(&stats[128 + col], shq[tid] + shq[tid + 128]);
    }
}

// ---------------------------------------------------------------------------
// BN normalize + affine + (optional) ELU
// ---------------------------------------------------------------------------
__global__ __launch_bounds__(256) void bn_apply(
    const float* __restrict__ h2, const float* __restrict__ stats,
    const float* __restrict__ gamma, const float* __restrict__ beta,
    float* __restrict__ out, int total, int do_elu)
{
    int idx = blockIdx.x * 256 + threadIdx.x;
    if (idx >= total) return;
    int col = idx & 127;
    const float invN = 1.0f / (float)N_NODES;
    float mean = stats[col] * invN;
    float var  = stats[128 + col] * invN - mean * mean;
    float v = (h2[idx] - mean) * rsqrtf(var + BN_EPS) * gamma[col] + beta[col];
    if (do_elu && v <= 0.f) v = expm1f(v);
    out[idx] = v;
}

// ---------------------------------------------------------------------------
// Launcher
// ---------------------------------------------------------------------------
extern "C" void kernel_launch(void* const* d_in, const int* in_sizes, int n_in,
                              void* d_out, int out_size, void* d_ws, size_t ws_size,
                              hipStream_t stream)
{
    const int*   x     = (const int*)  d_in[0];
    const int*   ei    = (const int*)  d_in[1];
    const int*   ea    = (const int*)  d_in[2];
    const float* ta    = (const float*)d_in[3];
    const float* td    = (const float*)d_in[4];
    const float* tf    = (const float*)d_in[5];
    const float* tc    = (const float*)d_in[6];
    const float* th_   = (const float*)d_in[7];
    const float* tn    = (const float*)d_in[8];
    const float* ti    = (const float*)d_in[9];
    const float* e1    = (const float*)d_in[10];
    const float* e2    = (const float*)d_in[11];
    const float* e3    = (const float*)d_in[12];
    const float* e4    = (const float*)d_in[13];
    const float* W1    = (const float*)d_in[14];
    const float* b1    = (const float*)d_in[15];
    const float* W2    = (const float*)d_in[16];
    const float* b2    = (const float*)d_in[17];
    const float* gamma = (const float*)d_in[18];
    const float* beta  = (const float*)d_in[19];
    float* out = (float*)d_out;

    // workspace carve-up (256 B aligned)
    char* ws = (char*)d_ws;
    size_t off = 0;
    auto carve = [&](size_t bytes) {
        void* p = ws + off;
        off = (off + bytes + 255) & ~(size_t)255;
        return p;
    };
    float*    h     = (float*)   carve((size_t)N_NODES * DFEAT * 4);
    float*    agg   = (float*)   carve((size_t)N_NODES * DFEAT * 4);
    float*    h2    = (float*)   carve((size_t)N_NODES * DFEAT * 4);
    _Float16* X16   = (_Float16*)carve((size_t)N_NODES * DFEAT * 2);
    _Float16* hid   = (_Float16*)carve((size_t)N_NODES * 256 * 2);
    _Float16* W1t   = (_Float16*)carve((size_t)NLAYERS * 256 * 128 * 2);
    _Float16* W2t   = (_Float16*)carve((size_t)NLAYERS * 128 * 256 * 2);
    float*    stats = (float*)   carve(256 * 4);

    const int EW_BLOCKS = (N_NODES * DFEAT) / 256;     // 50000
    const int PACK_BLOCKS = (2 * NLAYERS * 128 * 256) / 256;
    const int ROWBLOCKS = (N_NODES + 255) / 256;       // 391 (waves cover 32 rows)

    pack_weights<<<PACK_BLOCKS, 256, 0, stream>>>(W1, W2, W1t, W2t);
    node_embed<<<EW_BLOCKS, 256, 0, stream>>>(x, ta, td, tf, tc, th_, tn, ti, h);

    for (int l = 0; l < NLAYERS; ++l) {
        const float* t1 = e1 + (size_t)l * 7 * DFEAT;
        const float* t2 = e2 + (size_t)l * 3 * DFEAT;
        const float* t3 = e3 + (size_t)l * 3 * DFEAT;
        const float* t4 = e4 + (size_t)l * 6 * DFEAT;

        hipMemsetAsync(agg, 0, (size_t)N_NODES * DFEAT * 4, stream);
        hipMemsetAsync(stats, 0, 256 * 4, stream);

        scatter_msgs<<<N_EDGES / 8, 256, 0, stream>>>(h, ei, ea, t1, t2, t3, t4, agg);
        combine_x16<<<EW_BLOCKS, 256, 0, stream>>>(agg, h, t1, t2, t3, t4, X16);

        // 32x64 tile per wave: 8 WMMAs per k-step, each B feeds 2 WMMAs,
        // each A feeds 4; A re-read 4x, B LDS traffic halved.
        gemm_wmma<128, 256, 2, 4, true><<<dim3(ROWBLOCKS, 256 / 64), 256, 0, stream>>>(
            X16, W1t + (size_t)l * 256 * 128, b1 + (size_t)l * 256,
            hid, nullptr, N_NODES);

        gemm_wmma<256, 128, 2, 4, false><<<dim3(ROWBLOCKS, 128 / 64), 256, 0, stream>>>(
            hid, W2t + (size_t)l * 128 * 256, b2 + (size_t)l * 128,
            nullptr, h2, N_NODES);

        bn_stats<<<256, 256, 0, stream>>>(h2, stats, N_NODES);

        float* dstp = (l == NLAYERS - 1) ? out : h;
        bn_apply<<<EW_BLOCKS, 256, 0, stream>>>(
            h2, stats, gamma + (size_t)l * DFEAT, beta + (size_t)l * DFEAT,
            dstp, N_NODES * DFEAT, l != NLAYERS - 1);
    }
}
